// SCNN_12128987644087
// MI455X (gfx1250) — matrix operand order repaired and checked by reading, not compiled, not run on hardware
//
#include <hip/hip_runtime.h>
#include <hip/hip_bf16.h>
#include <stdint.h>

// ---------------- problem constants (from reference setup_inputs) ----------
#define DIN    784
#define H1DIM  2048
#define H2DIM  10
#define KMEM   10
#define UDIM   2832      // DIN + H1DIM
#define LDIM   128
#define BSZ    1024
#define TW     20        // time_window (scalar input; value fixed by setup_inputs)
#define KPAD   800       // 784 padded to multiple of 32
#define NWORDS 25        // 800 bits / 32 per (t,b) row
#define HALF_N 401408u   // (BSZ*DIN)/2 for threefry pairing
#define THRESHV 0.5f
#define DECAYV  0.2f

typedef __attribute__((ext_vector_type(16))) _Float16 v16h;
typedef __attribute__((ext_vector_type(8)))  float    v8f;
typedef __attribute__((ext_vector_type(4)))  unsigned int u32x4;

// ---------------- Threefry-2x32 (matches JAX threefry2x32) -----------------
__device__ __forceinline__ void tf_round(uint32_t& x0, uint32_t& x1, int r) {
  x0 += x1; x1 = (x1 << r) | (x1 >> (32 - r)); x1 ^= x0;
}
__device__ __forceinline__ void threefry2x32(uint32_t k0, uint32_t k1,
                                             uint32_t x0, uint32_t x1,
                                             uint32_t& o0, uint32_t& o1) {
  uint32_t k2 = k0 ^ k1 ^ 0x1BD11BDAu;
  x0 += k0; x1 += k1;
  tf_round(x0,x1,13); tf_round(x0,x1,15); tf_round(x0,x1,26); tf_round(x0,x1,6);
  x0 += k1; x1 += k2 + 1u;
  tf_round(x0,x1,17); tf_round(x0,x1,29); tf_round(x0,x1,16); tf_round(x0,x1,24);
  x0 += k2; x1 += k0 + 2u;
  tf_round(x0,x1,13); tf_round(x0,x1,15); tf_round(x0,x1,26); tf_round(x0,x1,6);
  x0 += k0; x1 += k1 + 3u;
  tf_round(x0,x1,17); tf_round(x0,x1,29); tf_round(x0,x1,16); tf_round(x0,x1,24);
  x0 += k1; x1 += k2 + 4u;
  tf_round(x0,x1,13); tf_round(x0,x1,15); tf_round(x0,x1,26); tf_round(x0,x1,6);
  x0 += k2; x1 += k0 + 5u;
  o0 = x0; o1 = x1;
}

// ---------------- WMMA fragment loads from LDS (row-major [row][k] f16) ----
// A-matrix 16x32 f16 layout (ISA 7.12.2): per lane (m = lane&15, h = lane>>4):
//   VGPR j=0..3 hold K = h*8 + {0..7};  j=4..7 hold K = 16 + h*8 + {0..7}
__device__ __forceinline__ v16h load_fragA(const _Float16* p, int row,
                                           int stride, int kBase, int hlf) {
  union { v16h h; u32x4 q[2]; } f;
  const _Float16* r = p + row * stride + kBase + hlf * 8;
  f.q[0] = *(const u32x4*)(r);
  f.q[1] = *(const u32x4*)(r + 16);
  return f.h;
}
// B-matrix 32x16 f16 layout (matches A): per lane (n = lane&15,
// h = lane>>4): e<8: K = h*8 + e; e>=8: K = 16 + h*8 + (e-8)
__device__ __forceinline__ v16h load_fragB(const _Float16* p, int row,
                                           int stride, int kBase, int hlf) {
  union { v16h h; u32x4 q[2]; } f;
  const _Float16* r = p + row * stride + kBase + hlf * 8;
  f.q[0] = *(const u32x4*)(r);
  f.q[1] = *(const u32x4*)(r + 16);
  return f.h;
}

// ---------------- kernel 1: zero y2 buffer + a1/a2/cs accumulators ---------
__global__ void __launch_bounds__(256) zero_kernel(float* y2, float* acc3) {
  int i = blockIdx.x * blockDim.x + threadIdx.x;
  if (i < TW * BSZ * H2DIM) y2[i] = 0.0f;
  if (i < 3) acc3[i] = 0.0f;
}

// ---------------- kernel 2: preprocess val1/val2 over core memory ----------
__global__ void __launch_bounds__(256) prep_kernel(const float* __restrict__ vec,
                                                   const int* __restrict__ cnt,
                                                   float* __restrict__ v1,
                                                   float* __restrict__ v2) {
  int u = blockIdx.x * blockDim.x + threadIdx.x;
  if (u >= UDIM) return;
  float a = 0.0f, b = 0.0f;
  for (int k = 0; k < KMEM; ++k) {
    const float* p = vec + ((size_t)k * UDIM + u) * LDIM;
    int li = -1;
    for (int l = 0; l < LDIM; ++l)
      if (p[l] != 0.0f) li = l;
    int c = cnt[k * UDIM + u];
    if (c > 0) {
      b += (float)c;
      if (li >= 0) a += (float)(li - 1);   // last_idx - 1 (can be -1)
    }
  }
  v1[u] = a; v2[u] = b;
}

// ---------------- kernel 3: Bernoulli sampling -> packed spike bits --------
// Also accumulates the x-side of a1/a2/cs: sum_{t,b,d} x * v{1,2}x[d], sum x.
__global__ void __launch_bounds__(256) sample_kernel(const float* __restrict__ input,
                                                     const float* __restrict__ v1,
                                                     const float* __restrict__ v2,
                                                     unsigned int* __restrict__ xbits,
                                                     float* __restrict__ acc3) {
  __shared__ float red[256];
  int tid = threadIdx.x;
  int gid = blockIdx.x * blockDim.x + tid;       // word index
  float s1 = 0.0f, s2 = 0.0f, sc = 0.0f;
  if (gid < TW * BSZ * NWORDS) {
    int w  = gid % NWORDS;
    int tb = gid / NWORDS;
    int b  = tb % BSZ;
    int t  = tb / BSZ;
    // fold_in(key(42), t): threefry((0,42), (0,t))
    uint32_t fk0, fk1;
    threefry2x32(0u, 42u, 0u, (uint32_t)t, fk0, fk1);
    unsigned int word = 0u;
    for (int bit = 0; bit < 32; ++bit) {
      int d = w * 32 + bit;
      if (d < DIN) {
        unsigned int i = (unsigned)(b * DIN + d);
        uint32_t o0, o1, r;
        if (i < HALF_N) { threefry2x32(fk0, fk1, i, i + HALF_N, o0, o1); r = o0; }
        else            { threefry2x32(fk0, fk1, i - HALF_N, i, o0, o1); r = o1; }
        float u = __uint_as_float((r >> 9) | 0x3f800000u) - 1.0f;
        if (input[b * DIN + d] > u) {
          word |= (1u << bit);
          s1 += v1[d]; s2 += v2[d]; sc += 1.0f;
        }
      }
    }
    xbits[gid] = word;
  }
  // block reductions -> 3 atomics per block
  red[tid] = s1; __syncthreads();
  for (int o = 128; o > 0; o >>= 1) { if (tid < o) red[tid] += red[tid + o]; __syncthreads(); }
  if (tid == 0) atomicAdd(&acc3[0], red[0]);
  __syncthreads();
  red[tid] = s2; __syncthreads();
  for (int o = 128; o > 0; o >>= 1) { if (tid < o) red[tid] += red[tid + o]; __syncthreads(); }
  if (tid == 0) atomicAdd(&acc3[1], red[0]);
  __syncthreads();
  red[tid] = sc; __syncthreads();
  for (int o = 128; o > 0; o >>= 1) { if (tid < o) red[tid] += red[tid + o]; __syncthreads(); }
  if (tid == 0) atomicAdd(&acc3[2], red[0]);
}

// ---------------- kernel 4: fused WMMA GEMM + h1 recurrence ----------------
// Grid (16, 32): block tile = 64 batch rows x 64 h1 cols, persistent over t.
// LDS (dynamic, ~218.5 KB of the WGP's 320 KB):
//   W1 tile  f16 [64][KPAD]  (100 KB, staged once)
//   X  tile  f16 [64][KPAD]  (100 KB, re-expanded per timestep)
//   Hs spike f32 [64][64]    (16 KB)
//   W2 tile  f32 [10][64]    (2.5 KB, staged once)
// Only 3 barriers per timestep; the 25-chunk WMMA loop is barrier-free.
__global__ void __launch_bounds__(256)
scnn_gemm_kernel(const float* __restrict__ W1, const float* __restrict__ b1,
                 const float* __restrict__ W2, const float* __restrict__ v1,
                 const float* __restrict__ v2,
                 const unsigned int* __restrict__ xbits,
                 float* __restrict__ y2, float* __restrict__ acc3) {
  extern __shared__ char smem[];
  _Float16* W1t = (_Float16*)smem;                         // [64][KPAD]
  _Float16* Xt  = (_Float16*)(smem + 64 * KPAD * 2);       // [64][KPAD]
  float*    Hs  = (float*)(smem + 2 * 64 * KPAD * 2);      // [64][64]
  float*    W2t = (float*)(smem + 2 * 64 * KPAD * 2 + 64 * 64 * 4); // [10][64]

  const int tid   = threadIdx.x;
  const int lane  = tid & 31;
  const int wave  = tid >> 5;
  const int hlf   = lane >> 4;         // lane half (0/1)
  const int l16   = lane & 15;
  const int waveM = wave & 1;          // batch 32-row group
  const int nBase = (wave >> 1) * 16;  // h1 16-col group
  const int bBase = blockIdx.x * 64;
  const int hBase = blockIdx.y * 64;

  // Stage this block's W1 tile into LDS once (f32 -> f16), zero-pad K>=784.
  for (int i = tid; i < 64 * KPAD; i += 256) {
    int n = i / KPAD, k = i - n * KPAD;
    float w = (k < DIN) ? W1[(hBase + n) * DIN + k] : 0.0f;
    W1t[i] = (_Float16)w;
  }
  // Stage W2 tile (10 x 64) once.
  for (int i = tid; i < H2DIM * 64; i += 256) {
    int j = i >> 6, hl = i & 63;
    W2t[i] = W2[j * H1DIM + hBase + hl];
  }

  const int   hG  = hBase + nBase + l16;   // this thread's h1 column (fixed)
  const float b1v = b1[hG];
  const float v1h = v1[DIN + hG];
  const float v2h = v2[DIN + hG];

  float h1m[2][8];
#pragma unroll
  for (int tt = 0; tt < 2; ++tt)
#pragma unroll
    for (int rr = 0; rr < 8; ++rr) h1m[tt][rr] = 0.0f;
  float s1 = 0.0f, s2 = 0.0f, sc = 0.0f;

  const int xrowIdx = tid >> 2;        // 4 threads cooperate per batch row
  const int xseg    = tid & 3;         // each covers 8 bits of every word

#pragma unroll 1
  for (int t = 0; t < TW; ++t) {
    // ---- expand the full spike tile for this timestep into LDS ----------
    __syncthreads();   // prior WMMA reads of Xt / y2 reads of Hs are done
    {
      const unsigned int* xrow =
          xbits + ((t * BSZ) + (bBase + xrowIdx)) * NWORDS;
      _Float16* xdst = Xt + xrowIdx * KPAD + xseg * 8;
#pragma unroll 5
      for (int c = 0; c < NWORDS; ++c) {
        unsigned int wb = xrow[c] >> (xseg * 8);
        u32x4 pk;
#pragma unroll
        for (int p2 = 0; p2 < 4; ++p2) {
          unsigned int lo = (wb >> (2 * p2)) & 1u;
          unsigned int hi = (wb >> (2 * p2 + 1)) & 1u;
          pk[p2] = (lo ? 0x00003C00u : 0u) | (hi ? 0x3C000000u : 0u);
        }
        *(u32x4*)(xdst + c * 32) = pk;   // bits 16..31 of word 24 are 0 => pad
      }
    }
    __syncthreads();

    // ---- barrier-free WMMA sweep over K = 25 chunks of 32 ---------------
    v8f acc0 = {}, acc1 = {};
#pragma unroll 5
    for (int c = 0; c < NWORDS; ++c) {
      const int kB = c * 32;
      v16h bf = load_fragB(W1t, nBase + l16, KPAD, kB, hlf);
      v16h a0 = load_fragA(Xt, waveM * 32 + l16, KPAD, kB, hlf);
      v16h a1 = load_fragA(Xt, waveM * 32 + 16 + l16, KPAD, kB, hlf);
      acc0 = __builtin_amdgcn_wmma_f32_16x16x32_f16(false, a0, false, bf,
                                                    (short)0, acc0, false, false);
      acc1 = __builtin_amdgcn_wmma_f32_16x16x32_f16(false, a1, false, bf,
                                                    (short)0, acc1, false, false);
    }

    // ---- membrane update: h1m = h1m*DECAY*(1-h1s_prev) + z + b1 ---------
    float cntv = 0.0f;
#pragma unroll
    for (int tt = 0; tt < 2; ++tt) {
#pragma unroll
      for (int rr = 0; rr < 8; ++rr) {
        float z = (tt == 0) ? acc0[rr] : acc1[rr];
        float m = h1m[tt][rr];
        float dec = (m > THRESHV) ? 0.0f : DECAYV;  // DECAY*(1 - prev spike)
        m = m * dec + z + b1v;
        h1m[tt][rr] = m;
        float sp = (m > THRESHV) ? 1.0f : 0.0f;
        cntv += sp;
        int mloc = waveM * 32 + tt * 16 + hlf * 8 + rr;   // C layout: m=r+8*half
        Hs[mloc * 64 + nBase + l16] = sp;
      }
    }
    s1 += cntv * v1h; s2 += cntv * v2h; sc += cntv;
    __syncthreads();

    // ---- partial h2 drive: y2[t,b,:] += Hs(64x64) @ W2t^T ---------------
    for (int p = tid; p < 64 * H2DIM; p += 256) {
      int bl = p / H2DIM, j = p - bl * H2DIM;
      float sum = 0.0f;
#pragma unroll 8
      for (int hl = 0; hl < 64; ++hl) sum += Hs[bl * 64 + hl] * W2t[j * 64 + hl];
      atomicAdd(&y2[((t * BSZ) + bBase + bl) * H2DIM + j], sum);
    }
  }

  // ---- reduce this block's a1/a2/cs h-side contributions ----------------
  float* red = Hs;   // reuse (barrier below orders vs. y2 readers)
  __syncthreads();
  red[tid] = s1; __syncthreads();
  for (int o = 128; o > 0; o >>= 1) { if (tid < o) red[tid] += red[tid + o]; __syncthreads(); }
  if (tid == 0) atomicAdd(&acc3[0], red[0]);
  __syncthreads();
  red[tid] = s2; __syncthreads();
  for (int o = 128; o > 0; o >>= 1) { if (tid < o) red[tid] += red[tid + o]; __syncthreads(); }
  if (tid == 0) atomicAdd(&acc3[1], red[0]);
  __syncthreads();
  red[tid] = sc; __syncthreads();
  for (int o = 128; o > 0; o >>= 1) { if (tid < o) red[tid] += red[tid + o]; __syncthreads(); }
  if (tid == 0) atomicAdd(&acc3[2], red[0]);
}

// ---------------- kernel 5: h2 recurrence + outputs (elementwise) ----------
__global__ void __launch_bounds__(256) finalize_kernel(const float* __restrict__ y2,
                                                       const float* __restrict__ b2,
                                                       float* __restrict__ out) {
  int idx = blockIdx.x * blockDim.x + threadIdx.x;
  if (idx >= BSZ * H2DIM) return;
  int b = idx / H2DIM, j = idx - b * H2DIM;
  float bj = b2[j];
  float m = 0.0f, s = 0.0f, sum = 0.0f;
#pragma unroll 1
  for (int t = 0; t < TW; ++t) {
    float y = y2[((t * BSZ) + b) * H2DIM + j] + bj;
    m = m * DECAYV * (1.0f - s) + y;
    s = (m > THRESHV) ? 1.0f : 0.0f;
    sum += s;
  }
  out[idx] = sum / (float)TW;
}

// ---------------- launch ---------------------------------------------------
extern "C" void kernel_launch(void* const* d_in, const int* in_sizes, int n_in,
                              void* d_out, int out_size, void* d_ws, size_t ws_size,
                              hipStream_t stream) {
  const float* input = (const float*)d_in[0];
  const float* W1    = (const float*)d_in[1];
  const float* b1    = (const float*)d_in[2];
  const float* W2    = (const float*)d_in[3];
  const float* b2    = (const float*)d_in[4];
  const float* cmv   = (const float*)d_in[5];
  const int*   cmc   = (const int*)d_in[6];
  // d_in[7] = time_window; fixed to TW=20 per setup_inputs (host cannot read
  // device scalars inside graph capture).
  (void)in_sizes; (void)n_in; (void)out_size; (void)ws_size;

  float* out = (float*)d_out;
  // workspace layout (~2.9 MB):
  float*        y2    = (float*)d_ws;                 // TW*BSZ*H2 = 204800 f
  float*        v1    = y2 + TW * BSZ * H2DIM;        // UDIM
  float*        v2    = v1 + UDIM;                    // UDIM
  unsigned int* xbits = (unsigned int*)(v2 + UDIM);   // TW*BSZ*NWORDS words
  float*        acc3  = out + BSZ * H2DIM;            // a1, a2, cs in d_out

  zero_kernel<<<(TW * BSZ * H2DIM + 255) / 256, 256, 0, stream>>>(y2, acc3);
  prep_kernel<<<(UDIM + 255) / 256, 256, 0, stream>>>(cmv, cmc, v1, v2);
  sample_kernel<<<(TW * BSZ * NWORDS + 255) / 256, 256, 0, stream>>>(
      input, v1, v2, xbits, acc3);

  dim3 grid(BSZ / 64, H1DIM / 64);
  size_t shbytes = (size_t)2 * 64 * KPAD * 2      // W1t + Xt
                 + (size_t)64 * 64 * 4            // Hs
                 + (size_t)H2DIM * 64 * 4;        // W2t
  scnn_gemm_kernel<<<grid, 256, shbytes, stream>>>(W1, b1, W2, v1, v2, xbits,
                                                   y2, acc3);

  finalize_kernel<<<(BSZ * H2DIM + 255) / 256, 256, 0, stream>>>(y2, b2, out);
}
